// Debias_v2_23897198035241
// MI455X (gfx1250) — compile-verified, matrix operand order checked by reading, not run on hardware
//
#include <hip/hip_runtime.h>
#include <hip/hip_bf16.h>
#include <math.h>

typedef __attribute__((ext_vector_type(16))) _Float16 v16h;
typedef __attribute__((ext_vector_type(8)))  float    v8f;

#define N_NODES  50000
#define IN_F     128
#define OUT_F    64
#define NIDX     5000
#define OMEGA_C  0.1f

__device__ __forceinline__ float lrelu(float v) { return v > 0.0f ? v : 0.01f * v; }

// Build an A/B fragment for v_wmma_f32_16x16x32_f16 from a contiguous fp32 row.
// Per ISA layout (16-bit A 16x32): lane half selects K base; halves j=0..7 -> K=k0..k0+7,
// j=8..15 -> K=k0+16..k0+23, with k0 = kk + (lane>=16 ? 8 : 0).
__device__ __forceinline__ v16h frag_contig_f32(const float* __restrict__ p, int k0) {
  v16h f;
#pragma unroll
  for (int j = 0; j < 8; ++j) f[j] = (_Float16)p[k0 + j];
#pragma unroll
  for (int j = 0; j < 8; ++j) f[j + 8] = (_Float16)p[k0 + 16 + j];
  return f;
}

// B fragment when B is row-major K x N (stride ldb), lane owns column `col`.
__device__ __forceinline__ v16h frag_strided_f32(const float* __restrict__ B, int k0, int col, int ldb) {
  v16h f;
#pragma unroll
  for (int j = 0; j < 8; ++j) f[j] = (_Float16)B[(k0 + j) * ldb + col];
#pragma unroll
  for (int j = 0; j < 8; ++j) f[j + 8] = (_Float16)B[(k0 + 16 + j) * ldb + col];
  return f;
}

// ---------------- degree sum (for Kthr) ----------------
__global__ void k_degsum(const int* __restrict__ degree, int* __restrict__ acc, int n) {
  int tid = blockIdx.x * blockDim.x + threadIdx.x;
  int local = 0;
  for (int i = tid; i < n; i += gridDim.x * blockDim.x) local += degree[i];
#pragma unroll
  for (int off = 16; off; off >>= 1) local += __shfl_xor(local, off, 32);
  if ((threadIdx.x & 31) == 0) atomicAdd(acc, local);
}

// ---------------- h = (x @ W^T + b) * 8 ----------------
// grid = N/16 blocks, 128 threads (4 waves); wave w owns 16x16 column tile w.
__global__ void k_gemm_h(const float* __restrict__ x, const float* __restrict__ W,
                         const float* __restrict__ bvec, float* __restrict__ h) {
  const int lane  = threadIdx.x & 31;
  const int wave  = threadIdx.x >> 5;
  const int rbase = blockIdx.x * 16;
  const int cbase = wave * 16;
  const int khalf = (lane >> 4) * 8;
  const int col   = cbase + (lane & 15);
  const float* xrow = x + (size_t)(rbase + (lane & 15)) * IN_F;
  const float* wrow = W + (size_t)col * IN_F;   // B[k][n] = W[n][k] -> row n of W
  v8f c = {};
#pragma unroll
  for (int kk = 0; kk < IN_F; kk += 32) {
    v16h a = frag_contig_f32(xrow, kk + khalf);
    v16h b = frag_contig_f32(wrow, kk + khalf);
    c = __builtin_amdgcn_wmma_f32_16x16x32_f16(false, a, false, b, (short)0, c, false, false);
  }
  const float bb = bvec[col];
#pragma unroll
  for (int r = 0; r < 8; ++r) {
    int row = rbase + r + ((lane >> 4) << 3);
    h[(size_t)row * OUT_F + col] = (c[r] + bb) * 8.0f;   // sqrt(DIM_M) = 8
  }
}

// ---------------- gamma/beta tables over 256 degrees ----------------
// grid = 16 blocks x 256 thr (8 waves): wave>>2 picks gamma/beta, wave&3 picks col tile.
__global__ void k_tables(const float* __restrict__ PE,
                         const float* __restrict__ Wg, const float* __restrict__ Wb,
                         const float* __restrict__ bg, const float* __restrict__ bb,
                         float* __restrict__ gtab, float* __restrict__ btab) {
  const int lane   = threadIdx.x & 31;
  const int wave   = threadIdx.x >> 5;
  const int matsel = wave >> 2;
  const int cbase  = (wave & 3) * 16;
  const int rbase  = blockIdx.x * 16;
  const float* Wm  = matsel ? Wb : Wg;
  const float* bm  = matsel ? bb : bg;
  float* out       = matsel ? btab : gtab;
  const int khalf  = (lane >> 4) * 8;
  const int col    = cbase + (lane & 15);
  const float* arow = PE + (size_t)(rbase + (lane & 15)) * OUT_F;  // PE: 256 x 64
  v8f c = {};
#pragma unroll
  for (int kk = 0; kk < 64; kk += 32) {
    v16h a = frag_contig_f32(arow, kk + khalf);
    v16h b = frag_strided_f32(Wm, kk + khalf, col, OUT_F);  // W_gamma is K x N row-major
    c = __builtin_amdgcn_wmma_f32_16x16x32_f16(false, a, false, b, (short)0, c, false, false);
  }
  const float bias = bm[col];
#pragma unroll
  for (int r = 0; r < 8; ++r) {
    int row = rbase + r + ((lane >> 4) << 3);
    out[row * OUT_F + col] = lrelu(c[r] + bias);
  }
}

// ---------------- per-node attention scalars s, t ----------------
__global__ void k_st(const float* __restrict__ h, const float* __restrict__ a,
                     float* __restrict__ s, float* __restrict__ t, int n) {
  int gw   = (blockIdx.x * blockDim.x + threadIdx.x) >> 5;
  int lane = threadIdx.x & 31;
  if (gw >= n) return;
  const float* hr = h + (size_t)gw * OUT_F;
  float h0 = hr[lane], h1 = hr[lane + 32];
  float sv = h0 * a[lane] + h1 * a[lane + 32];
  float tv = h0 * a[64 + lane] + h1 * a[96 + lane];
#pragma unroll
  for (int off = 16; off; off >>= 1) {
    sv += __shfl_xor(sv, off, 32);
    tv += __shfl_xor(tv, off, 32);
  }
  if (lane == 0) { s[gw] = sv; t[gw] = tv; }
}

// ---------------- fused edge pass: agg, e_rowsum, h_prime (L2 atomics) ----------------
__global__ void k_edges(const int* __restrict__ edge, int E,
                        const float* __restrict__ h, const float* __restrict__ s,
                        const float* __restrict__ t, float* __restrict__ agg,
                        float* __restrict__ hprime, float* __restrict__ erow) {
  const int lane = threadIdx.x & 31;
  const int gw   = (blockIdx.x * blockDim.x + threadIdx.x) >> 5;
  const int nw   = (gridDim.x * blockDim.x) >> 5;
  for (int e = gw; e < E; e += nw) {
    const int sv = edge[e];
    const int dv = edge[E + e];
    const float ee = __expf(-lrelu(s[sv] + t[dv]));
    const float* hd = h + (size_t)dv * OUT_F;
    const float h0 = hd[lane], h1 = hd[lane + 32];
    float* as = agg    + (size_t)sv * OUT_F;
    float* hs = hprime + (size_t)sv * OUT_F;
    atomicAdd(as + lane,      h0);
    atomicAdd(as + lane + 32, h1);
    atomicAdd(hs + lane,      ee * h0);
    atomicAdd(hs + lane + 32, ee * h1);
    if (lane == 0) atomicAdd(erow + sv, ee);
  }
}

// ---------------- fused FiLM GEMMs + bias + final output ----------------
__global__ void k_film(const float* __restrict__ agg, const int* __restrict__ degree,
                       const float* __restrict__ Wadd, const float* __restrict__ Wrev,
                       const float* __restrict__ gtab, const float* __restrict__ btab,
                       const float* __restrict__ hprime, const float* __restrict__ erow,
                       const int* __restrict__ degsum,
                       float* __restrict__ badd, float* __restrict__ brev,
                       float* __restrict__ out) {
  const int lane  = threadIdx.x & 31;
  const int wave  = threadIdx.x >> 5;
  const int rbase = blockIdx.x * 16;
  const int cbase = wave * 16;
  const float Kthr = (float)(*degsum) / (float)N_NODES;
  const int  arow = rbase + (lane & 15);
  const int  degA = degree[arow];
  const float inv = degA ? 1.0f / (float)degA : 0.0f;
  const float* ap = agg + (size_t)arow * OUT_F;
  const int khalf = (lane >> 4) * 8;
  const int col   = cbase + (lane & 15);
  const float* wa = Wadd + (size_t)col * OUT_F;  // i @ Wadd^T -> B[k][n] = Wadd[n][k]
  const float* wr = Wrev + (size_t)col * OUT_F;
  v8f cA = {}, cR = {};
#pragma unroll
  for (int kk = 0; kk < 64; kk += 32) {
    const int k0 = kk + khalf;
    v16h a;
#pragma unroll
    for (int j = 0; j < 8; ++j) a[j]     = (_Float16)(ap[k0 + j] * inv);
#pragma unroll
    for (int j = 0; j < 8; ++j) a[j + 8] = (_Float16)(ap[k0 + 16 + j] * inv);
    v16h bA = frag_contig_f32(wa, k0);
    v16h bR = frag_contig_f32(wr, k0);
    cA = __builtin_amdgcn_wmma_f32_16x16x32_f16(false, a, false, bA, (short)0, cA, false, false);
    cR = __builtin_amdgcn_wmma_f32_16x16x32_f16(false, a, false, bR, (short)0, cR, false, false);
  }
#pragma unroll
  for (int r = 0; r < 8; ++r) {
    const int row = rbase + r + ((lane >> 4) << 3);
    const int d   = degree[row];
    const float g  = gtab[d * OUT_F + col];
    const float be = btab[d * OUT_F + col];
    const float va = (g + 1.0f) * cA[r] + be;
    const float vr = (g + 1.0f) * cR[r] + be;
    const size_t o = (size_t)row * OUT_F + col;
    badd[o] = va;
    brev[o] = vr;
    const float R    = ((float)d < Kthr) ? 1.0f : 0.0f;
    const float bias = OMEGA_C * (R * va - (1.0f - R) * vr);
    const float ov   = (hprime[o] + bias) / (erow[row] + 1.00001f);
    out[o] = lrelu(ov);
  }
}

// ---------------- losses over idx sample ----------------
__global__ void k_loss(const int* __restrict__ idx, const int* __restrict__ degree,
                       const float* __restrict__ badd, const float* __restrict__ brev,
                       const float* __restrict__ gtab, const float* __restrict__ btab,
                       const int* __restrict__ degsum, float* __restrict__ acc) {
  const int lane = threadIdx.x & 31;
  const int gw   = (blockIdx.x * blockDim.x + threadIdx.x) >> 5;
  if (gw >= NIDX) return;
  const int v = idx[gw];
  const int d = degree[v];
  const float Kthr = (float)(*degsum) / (float)N_NODES;
  const float R = ((float)d < Kthr) ? 1.0f : 0.0f;
  const size_t o = (size_t)v * OUT_F;
  float a0 = badd[o + lane], a1 = badd[o + 32 + lane];
  float r0 = brev[o + lane], r1 = brev[o + 32 + lane];
  float g0 = gtab[d * OUT_F + lane], g1 = gtab[d * OUT_F + 32 + lane];
  float t0 = btab[d * OUT_F + lane], t1 = btab[d * OUT_F + 32 + lane];
  float sa = a0 * a0 + a1 * a1;
  float sr = r0 * r0 + r1 * r1;
  float sg = g0 * g0 + g1 * g1;
  float sb = t0 * t0 + t1 * t1;
#pragma unroll
  for (int off = 16; off; off >>= 1) {
    sa += __shfl_xor(sa, off, 32);
    sr += __shfl_xor(sr, off, 32);
    sg += __shfl_xor(sg, off, 32);
    sb += __shfl_xor(sb, off, 32);
  }
  if (lane == 0) {
    atomicAdd(acc + 0, R * sqrtf(sa) + (1.0f - R) * sqrtf(sr));
    atomicAdd(acc + 1, sqrtf(sg) + sqrtf(sb));
  }
}

__global__ void k_scalars(const float* __restrict__ acc, float* __restrict__ out) {
  if (threadIdx.x == 0 && blockIdx.x == 0) {
    out[(size_t)N_NODES * OUT_F]     = acc[0] / (float)NIDX;
    out[(size_t)N_NODES * OUT_F + 1] = acc[1] / (float)NIDX;
  }
}

extern "C" void kernel_launch(void* const* d_in, const int* in_sizes, int n_in,
                              void* d_out, int out_size, void* d_ws, size_t ws_size,
                              hipStream_t stream) {
  const float* x    = (const float*)d_in[0];
  const float* W    = (const float*)d_in[1];
  const float* bvec = (const float*)d_in[2];
  const float* a    = (const float*)d_in[3];
  const float* Wg   = (const float*)d_in[4];
  const float* Wb   = (const float*)d_in[5];
  const float* bg   = (const float*)d_in[6];
  const float* bb   = (const float*)d_in[7];
  const float* Wadd = (const float*)d_in[8];
  const float* Wrev = (const float*)d_in[9];
  const float* PE   = (const float*)d_in[10];
  const int*   edge = (const int*)d_in[11];
  const int*   degree = (const int*)d_in[12];
  const int*   idx  = (const int*)d_in[13];
  const int E = in_sizes[11] / 2;

  // workspace layout (floats)
  float* ws = (float*)d_ws;
  size_t o = 0;
  float* agg    = ws + o; o += (size_t)N_NODES * OUT_F;
  float* hprime = ws + o; o += (size_t)N_NODES * OUT_F;
  float* erow   = ws + o; o += N_NODES;
  float* acc    = ws + o; o += 4;           // [0]=L_b sum, [1]=L_film sum
  int*   degsum = (int*)(ws + o); o += 4;   // [0]=sum(degree)
  float* h      = ws + o; o += (size_t)N_NODES * OUT_F;
  float* sArr   = ws + o; o += N_NODES;
  float* tArr   = ws + o; o += N_NODES;
  float* badd   = ws + o; o += (size_t)N_NODES * OUT_F;
  float* brev   = ws + o; o += (size_t)N_NODES * OUT_F;
  float* gtab   = ws + o; o += 256 * OUT_F;
  float* btab   = ws + o; o += 256 * OUT_F;
  (void)o; (void)ws_size; (void)n_in; (void)out_size;

  // zero the accumulated region: agg, hprime, erow, acc, degsum (contiguous)
  hipMemsetAsync(ws, 0, ((size_t)2 * N_NODES * OUT_F + N_NODES + 8) * sizeof(float), stream);

  k_degsum<<<64, 256, 0, stream>>>(degree, degsum, N_NODES);
  k_gemm_h<<<N_NODES / 16, 128, 0, stream>>>(x, W, bvec, h);
  k_st<<<(N_NODES * 32 + 255) / 256, 256, 0, stream>>>(h, a, sArr, tArr, N_NODES);
  k_tables<<<16, 256, 0, stream>>>(PE, Wg, Wb, bg, bb, gtab, btab);
  k_edges<<<2048, 256, 0, stream>>>(edge, E, h, sArr, tArr, agg, hprime, erow);
  k_film<<<N_NODES / 16, 128, 0, stream>>>(agg, degree, Wadd, Wrev, gtab, btab,
                                           hprime, erow, degsum, badd, brev, (float*)d_out);
  k_loss<<<(NIDX * 32 + 255) / 256, 256, 0, stream>>>(idx, degree, badd, brev, gtab, btab, degsum, acc);
  k_scalars<<<1, 32, 0, stream>>>(acc, (float*)d_out);
}
